// Mask4Former_32263794328319
// MI455X (gfx1250) — matrix operand order repaired and checked by reading, not compiled, not run on hardware
//
#include <hip/hip_runtime.h>
#include <hip/hip_bf16.h>

// Problem dims
#define BB    4
#define QQ    100
#define NN    65536
#define NLL   16384
#define SS    8192
#define DD    128
#define FF    96
#define HH    8
#define HDD   16
#define DFFF  1024
#define NCLSS 21
#define NEGB  (-1000000000.0f)
#define SCALEV 0.25f   // 1/sqrt(16)

typedef __attribute__((ext_vector_type(16))) __bf16 bf16x16;
typedef __attribute__((ext_vector_type(8)))  float  f32x8;

__device__ __forceinline__ __bf16 f2bf(float f) {
  unsigned u = __builtin_bit_cast(unsigned, f);
  unsigned short h = (unsigned short)((u + 0x7FFFu + ((u >> 16) & 1u)) >> 16);
  return __builtin_bit_cast(__bf16, h);
}

__device__ __forceinline__ void cvt16(bf16x16& d, float4 a0, float4 a1,
                                      float4 a2, float4 a3) {
  d[0]=f2bf(a0.x);  d[1]=f2bf(a0.y);  d[2]=f2bf(a0.z);  d[3]=f2bf(a0.w);
  d[4]=f2bf(a1.x);  d[5]=f2bf(a1.y);  d[6]=f2bf(a1.z);  d[7]=f2bf(a1.w);
  d[8]=f2bf(a2.x);  d[9]=f2bf(a2.y);  d[10]=f2bf(a2.z); d[11]=f2bf(a2.w);
  d[12]=f2bf(a3.x); d[13]=f2bf(a3.y); d[14]=f2bf(a3.z); d[15]=f2bf(a3.w);
}

// ---------------------------------------------------------------------------
// bf16 WMMA GEMM:  out = act(A @ Wt^T + bias [+ res])
//   A:  (M,K) f32 row-major, optionally row-gathered
//   Wt: (N,K) f32 row-major (pre-transposed weights / mask_embed)
// Register blocking: each wave computes a 32x64 output block
//   (2 M-tiles x 4 N-tiles, 8 accumulators, up to 8 v_wmma per K-step).
// All tile-level control flow is SGPR-based (readfirstlane) so every WMMA
// executes with EXEC == all ones, as the ISA requires.
// ---------------------------------------------------------------------------
struct GemmP {
  const float* A; long long aB; int lda;
  const float* W; long long wB; int ldw;   // ldw == K
  const float* bias;
  const int* gat; long long gB;
  float* C; long long cB; int ldc;
  const float* res; long long rB;
  unsigned char* mk; long long mB; int ldm;
  int M, N, K, act; // act: 0 none, 1 relu, 2 sigmoid
};

__global__ __launch_bounds__(256) void gemm_wmma_bf16(GemmP p) {
  int b = blockIdx.y;
  const float* A = p.A + (long long)b * p.aB;
  const float* W = p.W + (long long)b * p.wB;
  const int* gat = p.gat ? (p.gat + (long long)b * p.gB) : (const int*)nullptr;
  float* C = p.C ? (p.C + (long long)b * p.cB) : (float*)nullptr;
  const float* res = p.res ? (p.res + (long long)b * p.rB) : (const float*)nullptr;
  unsigned char* mk = p.mk ? (p.mk + (long long)b * p.mB) : (unsigned char*)nullptr;

  int tilesM = (p.M + 15) >> 4;
  int tilesN = (p.N + 15) >> 4;
  int tilesM2 = (tilesM + 1) >> 1;
  int tilesN4 = (tilesN + 3) >> 2;
  // Force wave id into an SGPR: all tile control flow becomes scalar branches.
  int wave = __builtin_amdgcn_readfirstlane(blockIdx.x * 8 +
                                            ((int)threadIdx.x >> 5));
  if (wave >= tilesM2 * tilesN4) return;        // scalar branch
  int tm = wave / tilesN4;
  int tn = wave - tm * tilesN4;
  int mBase = tm << 5, nBase = tn << 6;
  int mt = tilesM - (tm << 1); if (mt > 2) mt = 2;  // SGPR
  int nt = tilesN - (tn << 2); if (nt > 4) nt = 4;  // SGPR

  int lane = (int)threadIdx.x & 31;
  int r    = lane & 15;
  int hs   = (lane >> 4) & 1;

  const float* Ar[2]; bool rOk[2]; bool fullM[2];
#pragma unroll
  for (int mi = 0; mi < 2; ++mi) {
    int row = mBase + 16 * mi + r;
    rOk[mi] = row < p.M;
    fullM[mi] = (mBase + 16 * mi + 16) <= p.M;   // scalar
    long long ar = 0;
    if (rOk[mi]) ar = gat ? (long long)gat[row] : (long long)row;
    Ar[mi] = A + ar * (long long)p.lda;
  }
  int colBase = nBase + r;

  f32x8 acc[2][4];
#pragma unroll
  for (int mi = 0; mi < 2; ++mi)
#pragma unroll
    for (int j = 0; j < 4; ++j)
      acc[mi][j] = (f32x8){0.f, 0.f, 0.f, 0.f, 0.f, 0.f, 0.f, 0.f};

  for (int kk = 0; kk < p.K; kk += 32) {
    // A fragments (one per M-tile), converted once, reused by 4 WMMAs each
    bf16x16 af[2];
#pragma unroll
    for (int mi = 0; mi < 2; ++mi) {
      if (mi < mt) {                             // scalar
        if (fullM[mi]) {                         // scalar fast path: no exec mask
          const float* ap = Ar[mi] + kk;
          __builtin_prefetch((const void*)(ap + 32), 0, 0);  // global_prefetch_b8
          const float4* p0 = (const float4*)(ap + hs * 8);
          const float4* p1 = (const float4*)(ap + 16 + hs * 8);
          cvt16(af[mi], p0[0], p0[1], p1[0], p1[1]);
        } else if (rOk[mi]) {                    // edge: per-lane guarded
          const float* ap = Ar[mi] + kk;
          const float4* p0 = (const float4*)(ap + hs * 8);
          const float4* p1 = (const float4*)(ap + 16 + hs * 8);
          cvt16(af[mi], p0[0], p0[1], p1[0], p1[1]);
        } else {
#pragma unroll
          for (int i = 0; i < 16; ++i) af[mi][i] = f2bf(0.f);
        }
      }
    }
#pragma unroll
    for (int j = 0; j < 4; ++j) {
      if (j < nt) {                              // scalar
        int col = colBase + 16 * j;
        bf16x16 bfv;
        if (nBase + 16 * j + 16 <= p.N) {        // scalar fast path
          const float* wb = W + (long long)col * p.ldw + kk;
          const float4* q0 = (const float4*)(wb + hs * 8);
          const float4* q1 = (const float4*)(wb + 16 + hs * 8);
          cvt16(bfv, q0[0], q0[1], q1[0], q1[1]);
        } else if (col < p.N) {                  // edge: per-lane guarded
          const float* wb = W + (long long)col * p.ldw + kk;
          const float4* q0 = (const float4*)(wb + hs * 8);
          const float4* q1 = (const float4*)(wb + 16 + hs * 8);
          cvt16(bfv, q0[0], q0[1], q1[0], q1[1]);
        } else {
#pragma unroll
          for (int i = 0; i < 16; ++i) bfv[i] = f2bf(0.f);
        }
#pragma unroll
        for (int mi = 0; mi < 2; ++mi) {
          if (mi < mt)                           // scalar: EXEC untouched
            acc[mi][j] = __builtin_amdgcn_wmma_f32_16x16x32_bf16(
                false, af[mi], false, bfv, (short)0, acc[mi][j], false, false);
        }
      }
    }
  }

  // Store: D layout — element i of v8f is row (tileBase + i + hs*8), col fixed
#pragma unroll
  for (int j = 0; j < 4; ++j) {
    if (j >= nt) continue;                       // scalar
    int col = colBase + 16 * j;
    if (col >= p.N) continue;                    // per-lane (stores only)
    float bv = p.bias ? p.bias[col] : 0.f;
#pragma unroll
    for (int mi = 0; mi < 2; ++mi) {
      if (mi >= mt) continue;                    // scalar
#pragma unroll
      for (int i = 0; i < 8; ++i) {
        int row = mBase + 16 * mi + i + hs * 8;
        if (row >= p.M) continue;
        float v = acc[mi][j][i] + bv;
        if (res) v += res[(long long)row * p.ldc + col];
        if (p.act == 1)      v = v > 0.f ? v : 0.f;
        else if (p.act == 2) v = 1.f / (1.f + __expf(-v));
        if (mk) mk[(long long)row * p.ldm + col] = (v < 0.f) ? (unsigned char)1
                                                             : (unsigned char)0;
        else    C[(long long)row * p.ldc + col] = v;
      }
    }
  }
}

// Transpose weights once: Wt[n*K + k] = W[k*N + n]
__global__ __launch_bounds__(256) void transpose_k(const float* W, float* Wt,
                                                   int K, int N) {
  int n = blockIdx.x * 256 + (int)threadIdx.x;
  int k = blockIdx.y;
  if (n < N) Wt[(long long)n * K + k] = W[(long long)k * N + n];
}

// ---------------------------------------------------------------------------
// LayerNorm over D=128 per row (one block per row, 128 threads)
// ---------------------------------------------------------------------------
__global__ __launch_bounds__(128) void ln_k(const float* x, const float* res,
                                            const float* g, const float* bb,
                                            float* out, int D) {
  int row = blockIdx.x, t = threadIdx.x;
  float v = x[(long long)row * D + t];
  if (res) v += res[(long long)row * D + t];
  __shared__ float sm[128];
  sm[t] = v; __syncthreads();
  for (int o = 64; o > 0; o >>= 1) { if (t < o) sm[t] += sm[t + o]; __syncthreads(); }
  float mean = sm[0] / (float)D; __syncthreads();
  float c = v - mean;
  sm[t] = c * c; __syncthreads();
  for (int o = 64; o > 0; o >>= 1) { if (t < o) sm[t] += sm[t + o]; __syncthreads(); }
  float var = sm[0] / (float)D;
  out[(long long)row * D + t] = c * rsqrtf(var + 1e-5f) * g[t] + bb[t];
}

__global__ void add_k(const float* a, const float* b, float* c, long long n) {
  long long i = (long long)blockIdx.x * blockDim.x + threadIdx.x;
  long long st = (long long)gridDim.x * blockDim.x;
  for (; i < n; i += st) c[i] = a[i] + b[i];
}

// gather b_pos (float, D) and b_attn (u8, Q) via rand_idx
__global__ __launch_bounds__(128) void gather_k(const float* pos, const unsigned char* amask,
                                                const int* idx, float* bpos,
                                                unsigned char* battn) {
  int b = blockIdx.y, s = blockIdx.x, t = threadIdx.x;
  int id = idx[(long long)b * SS + s];
  bpos[(((long long)b * SS + s)) * DD + t] = pos[(((long long)b * NLL + id)) * DD + t];
  if (t < QQ)
    battn[(((long long)b * SS + s)) * QQ + t] = amask[(((long long)b * NLL + id)) * QQ + t];
}

// all_masked[b,q] = AND over s of b_attn[b,s,q]
__global__ __launch_bounds__(256) void allmask_k(const unsigned char* battn,
                                                 unsigned char* allm) {
  int b = blockIdx.y, q = blockIdx.x, t = threadIdx.x;
  int acc = 1;
  for (int s = t; s < SS; s += 256) acc &= (int)battn[(((long long)b * SS + s)) * QQ + q];
  __shared__ int sm[256];
  sm[t] = acc; __syncthreads();
  for (int o = 128; o > 0; o >>= 1) { if (t < o) sm[t] &= sm[t + o]; __syncthreads(); }
  if (t == 0) allm[b * QQ + q] = (unsigned char)sm[0];
}

// ---------------------------------------------------------------------------
// Multi-head attention (softmax + AV) for one (b, query) per block.
// Q/K/V already projected (+bias). Optional additive mask bias for cross-attn.
// Two-pass max/sum per head; heads looped; K/V re-reads hit L2 (34MB << 192MB).
// ---------------------------------------------------------------------------
__global__ __launch_bounds__(256) void attn_k(const float* Qp, const float* Kp,
                                              const float* Vp, float* O,
                                              const unsigned char* battn,
                                              const unsigned char* allm,
                                              const unsigned char* maskidx,
                                              int Lq, int Lk) {
  int b = blockIdx.y, qi = blockIdx.x, t = threadIdx.x;
  __shared__ float qs[DD], osh[DD];
  __shared__ float red[256];
  __shared__ float accb[HDD * 256];
  if (t < DD) qs[t] = Qp[(((long long)b * Lq + qi)) * DD + t];
  __syncthreads();
  int amFlag = (battn && allm) ? (int)allm[b * Lq + qi] : 0;

  for (int h = 0; h < HH; ++h) {
    const float* qh = qs + h * HDD;
    // pass 1: max
    float mloc = -3.0e38f;
    for (int s = t; s < Lk; s += 256) {
      const float* kr = Kp + (((long long)b * Lk + s)) * DD + h * HDD;
      float sc = 0.f;
#pragma unroll
      for (int d = 0; d < HDD; ++d) sc += qh[d] * kr[d];
      sc *= SCALEV;
      if (battn) {
        int mbit = ((int)battn[(((long long)b * Lk + s)) * QQ + qi] && !amFlag) ||
                   (maskidx && (int)maskidx[(long long)b * Lk + s]);
        if (mbit) sc += NEGB;
      }
      mloc = fmaxf(mloc, sc);
    }
    red[t] = mloc; __syncthreads();
    for (int o = 128; o > 0; o >>= 1) { if (t < o) red[t] = fmaxf(red[t], red[t + o]); __syncthreads(); }
    float m = red[0]; __syncthreads();

    // pass 2: sum + weighted V accumulate
    float lloc = 0.f;
    float acc[HDD];
#pragma unroll
    for (int d = 0; d < HDD; ++d) acc[d] = 0.f;
    for (int s = t; s < Lk; s += 256) {
      const float* kr = Kp + (((long long)b * Lk + s)) * DD + h * HDD;
      float sc = 0.f;
#pragma unroll
      for (int d = 0; d < HDD; ++d) sc += qh[d] * kr[d];
      sc *= SCALEV;
      if (battn) {
        int mbit = ((int)battn[(((long long)b * Lk + s)) * QQ + qi] && !amFlag) ||
                   (maskidx && (int)maskidx[(long long)b * Lk + s]);
        if (mbit) sc += NEGB;
      }
      float pw = __expf(sc - m);
      lloc += pw;
      const float* vr = Vp + (((long long)b * Lk + s)) * DD + h * HDD;
#pragma unroll
      for (int d = 0; d < HDD; ++d) acc[d] += pw * vr[d];
    }
    red[t] = lloc; __syncthreads();
    for (int o = 128; o > 0; o >>= 1) { if (t < o) red[t] += red[t + o]; __syncthreads(); }
    float l = red[0]; __syncthreads();
#pragma unroll
    for (int d = 0; d < HDD; ++d) accb[d * 256 + t] = acc[d];
    __syncthreads();
    if (t < HDD) {
      float ssum = 0.f;
      for (int j = 0; j < 256; ++j) ssum += accb[t * 256 + j];
      osh[h * HDD + t] = ssum / l;
    }
    __syncthreads();
  }
  if (t < DD) O[(((long long)b * Lq + qi)) * DD + t] = osh[t];
}

// ---------------------------------------------------------------------------
// Host orchestration
// ---------------------------------------------------------------------------
static inline void gemm(hipStream_t st, int batches,
                        const float* A, long long aB, int lda,
                        const float* Wt, long long wB, int ldw,
                        const float* bias, const int* gat, long long gB,
                        float* C, long long cB, int ldc,
                        const float* res, long long rB,
                        unsigned char* mk, long long mB, int ldm,
                        int M, int N, int K, int act) {
  GemmP p;
  p.A = A; p.aB = aB; p.lda = lda;
  p.W = Wt; p.wB = wB; p.ldw = ldw;
  p.bias = bias; p.gat = gat; p.gB = gB;
  p.C = C; p.cB = cB; p.ldc = ldc;
  p.res = res; p.rB = rB;
  p.mk = mk; p.mB = mB; p.ldm = ldm;
  p.M = M; p.N = N; p.K = K; p.act = act;
  int tiles = ((((M + 15) / 16) + 1) / 2) * ((((N + 15) / 16) + 3) / 4);
  dim3 grid((unsigned)((tiles + 7) / 8), (unsigned)batches);
  gemm_wmma_bf16<<<grid, 256, 0, st>>>(p);
}

static inline void transpose(hipStream_t st, const float* W, float* Wt, int K, int N) {
  dim3 grid((unsigned)((N + 255) / 256), (unsigned)K);
  transpose_k<<<grid, 256, 0, st>>>(W, Wt, K, N);
}

extern "C" void kernel_launch(void* const* d_in, const int* in_sizes, int n_in,
                              void* d_out, int out_size, void* d_ws, size_t ws_size,
                              hipStream_t stream) {
  (void)in_sizes; (void)n_in; (void)out_size; (void)ws_size;
  const float* queries          = (const float*)d_in[0];
  const float* query_pos        = (const float*)d_in[1];
  const float* point_feats      = (const float*)d_in[2];
  const float* level_feats      = (const float*)d_in[3];
  const float* level_mask_feats = (const float*)d_in[4];
  const float* pos_enc_level    = (const float*)d_in[5];
  const int*   rand_idx         = (const int*)d_in[6];
  const unsigned char* mask_idx = (const unsigned char*)d_in[7];
  const float* ln_dec_g = (const float*)d_in[8],  * ln_dec_b = (const float*)d_in[9];
  const float* me_w1 = (const float*)d_in[10], * me_b1 = (const float*)d_in[11];
  const float* me_w2 = (const float*)d_in[12], * me_b2 = (const float*)d_in[13];
  const float* cls_w = (const float*)d_in[14], * cls_b = (const float*)d_in[15];
  const float* bb_w1 = (const float*)d_in[16], * bb_b1 = (const float*)d_in[17];
  const float* bb_w2 = (const float*)d_in[18], * bb_b2 = (const float*)d_in[19];
  const float* bb_w3 = (const float*)d_in[20], * bb_b3 = (const float*)d_in[21];
  const float* sq_w  = (const float*)d_in[22], * sq_b  = (const float*)d_in[23];
  const float* ffn_w1 = (const float*)d_in[24], * ffn_b1 = (const float*)d_in[25];
  const float* ffn_w2 = (const float*)d_in[26], * ffn_b2 = (const float*)d_in[27];
  const float* ffn_ln_g = (const float*)d_in[28], * ffn_ln_b = (const float*)d_in[29];
  const float* ca_wq = (const float*)d_in[30], * ca_wk = (const float*)d_in[31];
  const float* ca_wv = (const float*)d_in[32], * ca_wo = (const float*)d_in[33];
  const float* ca_bq = (const float*)d_in[34], * ca_bk = (const float*)d_in[35];
  const float* ca_bv = (const float*)d_in[36], * ca_bo = (const float*)d_in[37];
  const float* ca_ln_g = (const float*)d_in[38], * ca_ln_b = (const float*)d_in[39];
  const float* sa_wq = (const float*)d_in[40], * sa_wk = (const float*)d_in[41];
  const float* sa_wv = (const float*)d_in[42], * sa_wo = (const float*)d_in[43];
  const float* sa_bq = (const float*)d_in[44], * sa_bk = (const float*)d_in[45];
  const float* sa_bv = (const float*)d_in[46], * sa_bo = (const float*)d_in[47];
  const float* sa_ln_g = (const float*)d_in[48], * sa_ln_b = (const float*)d_in[49];

  float* out = (float*)d_out;
  float* out_x     = out;                       // (B,Q,D)   51200
  float* out_class = out + 51200;               // (B,Q,21)   8400
  float* out_bbox  = out + 51200 + 8400;        // (B,Q,6)    2400
  float* out_mask  = out + 51200 + 8400 + 2400; // (B,N,Q)

  char* ws = (char*)d_ws;
  const size_t SZ = 4ll * QQ * DD * sizeof(float); // 204800
  float* qn     = (float*)(ws + 0);
  float* qin    = (float*)(ws + 1 * SZ);
  float* qca    = (float*)(ws + 2 * SZ);
  float* membed = (float*)(ws + 3 * SZ);
  float* tmpA   = (float*)(ws + 4 * SZ);          // 400*1024*4 = 1638400
  float* tmpB   = (float*)(ws + 4 * SZ + 1638400);
  float* oca    = (float*)(ws + 5 * SZ + 1638400);
  float* x1     = (float*)(ws + 6 * SZ + 1638400);
  float* x1n    = (float*)(ws + 7 * SZ + 1638400);
  float* xq     = (float*)(ws + 8 * SZ + 1638400);
  float* qsa    = (float*)(ws + 9 * SZ + 1638400);
  float* ksa    = (float*)(ws + 10 * SZ + 1638400);
  float* vsa    = (float*)(ws + 11 * SZ + 1638400);
  float* osa    = (float*)(ws + 12 * SZ + 1638400);
  float* x2     = (float*)(ws + 13 * SZ + 1638400);
  float* x2n    = (float*)(ws + 14 * SZ + 1638400);
  float* fout   = (float*)(ws + 15 * SZ + 1638400);
  size_t off = 16 * SZ + 1638400;                 // 4915200
  unsigned char* allm  = (unsigned char*)(ws + off);          off += 1024;
  unsigned char* amask = (unsigned char*)(ws + off);          off += (size_t)BB * NLL * QQ; // 6553600
  unsigned char* battn = (unsigned char*)(ws + off);          off += (size_t)BB * SS * QQ;  // 3276800
  float* bpos = (float*)(ws + off);  off += (size_t)BB * SS * DD * 4;  // kin reuses this
  float* src  = (float*)(ws + off);  off += (size_t)BB * SS * DD * 4;
  float* kca  = (float*)(ws + off);  off += (size_t)BB * SS * DD * 4;
  float* vca  = (float*)(ws + off);  off += (size_t)BB * SS * DD * 4;
  // Pre-transposed weights (N,K layout)
  float* wt_me1 = (float*)(ws + off); off += 65536 * 4;
  float* wt_me2 = (float*)(ws + off); off += 65536 * 4;
  float* wt_bb1 = (float*)(ws + off); off += 65536 * 4;
  float* wt_bb2 = (float*)(ws + off); off += 65536 * 4;
  float* wt_caq = (float*)(ws + off); off += 65536 * 4;
  float* wt_cak = (float*)(ws + off); off += 65536 * 4;
  float* wt_cav = (float*)(ws + off); off += 65536 * 4;
  float* wt_cao = (float*)(ws + off); off += 65536 * 4;
  float* wt_saq = (float*)(ws + off); off += 65536 * 4;
  float* wt_sak = (float*)(ws + off); off += 65536 * 4;
  float* wt_sav = (float*)(ws + off); off += 65536 * 4;
  float* wt_sao = (float*)(ws + off); off += 65536 * 4;
  float* wt_cls = (float*)(ws + off); off += 12288;          // 21*128 floats
  float* wt_bb3 = (float*)(ws + off); off += 4096;           // 6*128 floats
  float* wt_sq  = (float*)(ws + off); off += 49152;          // 128*96 floats
  float* wt_f1  = (float*)(ws + off); off += 131072 * 4;     // 1024*128
  float* wt_f2  = (float*)(ws + off); off += 131072 * 4;     // 128*1024

  // One-time weight transposes (K,N)->(N,K)
  transpose(stream, me_w1, wt_me1, DD, DD);
  transpose(stream, me_w2, wt_me2, DD, DD);
  transpose(stream, bb_w1, wt_bb1, DD, DD);
  transpose(stream, bb_w2, wt_bb2, DD, DD);
  transpose(stream, ca_wq, wt_caq, DD, DD);
  transpose(stream, ca_wk, wt_cak, DD, DD);
  transpose(stream, ca_wv, wt_cav, DD, DD);
  transpose(stream, ca_wo, wt_cao, DD, DD);
  transpose(stream, sa_wq, wt_saq, DD, DD);
  transpose(stream, sa_wk, wt_sak, DD, DD);
  transpose(stream, sa_wv, wt_sav, DD, DD);
  transpose(stream, sa_wo, wt_sao, DD, DD);
  transpose(stream, cls_w, wt_cls, DD, NCLSS);
  transpose(stream, bb_w3, wt_bb3, DD, 6);
  transpose(stream, sq_w,  wt_sq,  FF, DD);
  transpose(stream, ffn_w1, wt_f1, DD, DFFF);
  transpose(stream, ffn_w2, wt_f2, DFFF, DD);

  const int MQ = BB * QQ; // 400 flattened rows, weights shared across batch

  // qin = queries + query_pos ; qn = LN(queries)
  add_k<<<200, 256, 0, stream>>>(queries, query_pos, qin, (long long)MQ * DD);
  ln_k<<<MQ, 128, 0, stream>>>(queries, nullptr, ln_dec_g, ln_dec_b, qn, DD);

  // mask_embed = relu(qn@me_w1+b1)@me_w2+b2
  gemm(stream, 1, qn, 0, DD, wt_me1, 0, DD, me_b1, nullptr, 0,
       tmpA, 0, DD, nullptr, 0, nullptr, 0, 0, MQ, DD, DD, 1);
  gemm(stream, 1, tmpA, 0, DD, wt_me2, 0, DD, me_b2, nullptr, 0,
       membed, 0, DD, nullptr, 0, nullptr, 0, 0, MQ, DD, DD, 0);
  // out_class
  gemm(stream, 1, qn, 0, DD, wt_cls, 0, DD, cls_b, nullptr, 0,
       out_class, 0, NCLSS, nullptr, 0, nullptr, 0, 0, MQ, NCLSS, DD, 0);
  // bbox MLP
  gemm(stream, 1, qn, 0, DD, wt_bb1, 0, DD, bb_b1, nullptr, 0,
       tmpA, 0, DD, nullptr, 0, nullptr, 0, 0, MQ, DD, DD, 1);
  gemm(stream, 1, tmpA, 0, DD, wt_bb2, 0, DD, bb_b2, nullptr, 0,
       tmpB, 0, DD, nullptr, 0, nullptr, 0, 0, MQ, DD, DD, 1);
  gemm(stream, 1, tmpB, 0, DD, wt_bb3, 0, DD, bb_b3, nullptr, 0,
       out_bbox, 0, 6, nullptr, 0, nullptr, 0, 0, MQ, 6, DD, 2);

  // out_mask[b,n,q] = point_feats[b,n,:] . mask_embed[b,q,:]   (Wt = mask_embed)
  gemm(stream, BB, point_feats, (long long)NN * DD, DD,
       membed, (long long)QQ * DD, DD, nullptr, nullptr, 0,
       out_mask, (long long)NN * QQ, QQ, nullptr, 0, nullptr, 0, 0,
       NN, QQ, DD, 0);
  // attn_mask (u8: logits<0)
  gemm(stream, BB, level_mask_feats, (long long)NLL * DD, DD,
       membed, (long long)QQ * DD, DD, nullptr, nullptr, 0,
       nullptr, 0, 0, nullptr, 0, amask, (long long)NLL * QQ, QQ,
       NLL, QQ, DD, 0);

  // gather b_pos / b_attn ; all_masked
  { dim3 g(SS, BB); gather_k<<<g, 128, 0, stream>>>(pos_enc_level, amask, rand_idx, bpos, battn); }
  { dim3 g(QQ, BB); allmask_k<<<g, 256, 0, stream>>>(battn, allm); }

  // src = gathered(level_feats) @ sq_w + sq_b
  gemm(stream, BB, level_feats, (long long)NLL * FF, FF,
       wt_sq, 0, FF, sq_b, rand_idx, SS,
       src, (long long)SS * DD, DD, nullptr, 0, nullptr, 0, 0,
       SS, DD, FF, 0);
  // kin = src + b_pos (in-place into bpos)
  add_k<<<4096, 256, 0, stream>>>(src, bpos, bpos, (long long)BB * SS * DD);

  // K/V/Q projections for cross-attn (flattened M)
  gemm(stream, 1, bpos, 0, DD, wt_cak, 0, DD, ca_bk, nullptr, 0,
       kca, 0, DD, nullptr, 0, nullptr, 0, 0, BB * SS, DD, DD, 0);
  gemm(stream, 1, src, 0, DD, wt_cav, 0, DD, ca_bv, nullptr, 0,
       vca, 0, DD, nullptr, 0, nullptr, 0, 0, BB * SS, DD, DD, 0);
  gemm(stream, 1, qin, 0, DD, wt_caq, 0, DD, ca_bq, nullptr, 0,
       qca, 0, DD, nullptr, 0, nullptr, 0, 0, MQ, DD, DD, 0);

  // cross-attention
  { dim3 g(QQ, BB); attn_k<<<g, 256, 0, stream>>>(qca, kca, vca, oca, battn, allm, mask_idx, QQ, SS); }
  // x1 = oca@wo + bo + queries ; x1n = LN(x1)
  gemm(stream, 1, oca, 0, DD, wt_cao, 0, DD, ca_bo, nullptr, 0,
       x1, 0, DD, queries, 0, nullptr, 0, 0, MQ, DD, DD, 0);
  ln_k<<<MQ, 128, 0, stream>>>(x1, nullptr, ca_ln_g, ca_ln_b, x1n, DD);

  // self-attention
  add_k<<<200, 256, 0, stream>>>(x1n, query_pos, xq, (long long)MQ * DD);
  gemm(stream, 1, xq, 0, DD, wt_saq, 0, DD, sa_bq, nullptr, 0,
       qsa, 0, DD, nullptr, 0, nullptr, 0, 0, MQ, DD, DD, 0);
  gemm(stream, 1, xq, 0, DD, wt_sak, 0, DD, sa_bk, nullptr, 0,
       ksa, 0, DD, nullptr, 0, nullptr, 0, 0, MQ, DD, DD, 0);
  gemm(stream, 1, x1n, 0, DD, wt_sav, 0, DD, sa_bv, nullptr, 0,
       vsa, 0, DD, nullptr, 0, nullptr, 0, 0, MQ, DD, DD, 0);
  { dim3 g(QQ, BB); attn_k<<<g, 256, 0, stream>>>(qsa, ksa, vsa, osa, nullptr, nullptr, nullptr, QQ, QQ); }
  gemm(stream, 1, osa, 0, DD, wt_sao, 0, DD, sa_bo, nullptr, 0,
       x2, 0, DD, x1n, 0, nullptr, 0, 0, MQ, DD, DD, 0);
  ln_k<<<MQ, 128, 0, stream>>>(x2, nullptr, sa_ln_g, sa_ln_b, x2n, DD);

  // FFN + final LN -> out_x
  gemm(stream, 1, x2n, 0, DD, wt_f1, 0, DD, ffn_b1, nullptr, 0,
       tmpA, 0, DFFF, nullptr, 0, nullptr, 0, 0, MQ, DFFF, DD, 1);
  gemm(stream, 1, tmpA, 0, DFFF, wt_f2, 0, DFFF, ffn_b2, nullptr, 0,
       fout, 0, DD, x2n, 0, nullptr, 0, 0, MQ, DD, DFFF, 0);
  ln_k<<<MQ, 128, 0, stream>>>(fout, nullptr, ffn_ln_g, ffn_ln_b, out_x, DD);
}
